// MoELayer_27530740367430
// MI455X (gfx1250) — matrix operand order repaired and checked by reading, not compiled, not run on hardware
//
#include <hip/hip_runtime.h>
#include <hip/hip_bf16.h>
#include <math.h>

// ---------------- problem constants (from reference) ----------------
#define TOP_K       2
#define NUM_EXPERTS 8
#define D_IN        2048
#define D_HID       8192
#define D_OUT       2048
#define BATCH       8192

// GEMM tiling
#define BM     256            // rows per block (8 waves x 2 subtiles x 16)
#define BN     64             // cols per block (4 n-tiles x 16)
#define KSTEP  32             // bf16 WMMA K depth
#define BK     64             // K depth per LDS buffer (2 WMMA substeps)
#define KPAD   72             // LDS row stride in bf16 elems (144B, 16B aligned)

typedef __attribute__((ext_vector_type(16))) __bf16 v16bf;
typedef __attribute__((ext_vector_type(8)))  float  v8f;

// ---------------- CDNA5 async copy to LDS (gfx1250, ASYNCcnt) ----------------
__device__ __forceinline__ void async_ld_b128(unsigned lds_off, const __bf16* g) {
    asm volatile("global_load_async_to_lds_b128 %0, %1, off"
                 :: "v"(lds_off), "v"((unsigned long long)(size_t)g)
                 : "memory");
}
__device__ __forceinline__ void wait_async0() {
    asm volatile("s_wait_asynccnt 0x0" ::: "memory");
}

// ---------------- WMMA helpers ----------------
__device__ __forceinline__ v8f wmma_bf16(v16bf a, v16bf b, v8f c) {
    return __builtin_amdgcn_wmma_f32_16x16x32_bf16(
        false, a, false, b, (short)0, c, false, false);
}

// A fragment (16x32 bf16): p already offset by lane-half (hi? +8 : +0)
__device__ __forceinline__ v16bf load_a_frag(const __bf16* __restrict__ p, int k0) {
    v16bf a{};
#pragma unroll
    for (int i = 0; i < 8; ++i) {
        a[i]     = p[k0 + i];          // K = k0 + hi*8 + i
        a[8 + i] = p[k0 + 16 + i];     // K = k0 + hi*8 + 16 + i
    }
    return a;
}

// B fragment (32x16 bf16) from LDS (bp -> row base + substep + hi*8)
__device__ __forceinline__ v16bf load_b_frag(const __bf16* bp) {
    v16bf b{};
#pragma unroll
    for (int i = 0; i < 16; ++i) b[i] = bp[(i & 7) + ((i >> 3) << 4)];     // K = k0 + hi*8 + (i<8 ? i : 16 + i-8)
    return b;
}

// ---------------- small utility kernels ----------------
__global__ void zero_f32_kernel(float* __restrict__ p, long long n) {
    long long i = (long long)blockIdx.x * blockDim.x + threadIdx.x;
    long long stride = (long long)gridDim.x * blockDim.x;
    for (; i < n; i += stride) p[i] = 0.0f;
}

__global__ void zero_i32_kernel(int* __restrict__ p, int n) {
    int i = blockIdx.x * blockDim.x + threadIdx.x;
    if (i < n) p[i] = 0;
}

__global__ void cvt_f32_bf16_kernel(const float* __restrict__ src,
                                    __bf16* __restrict__ dst, long long n) {
    long long i = (long long)blockIdx.x * blockDim.x + threadIdx.x;
    long long stride = (long long)gridDim.x * blockDim.x;
    for (; i < n; i += stride) dst[i] = (__bf16)src[i];
}

__global__ void scan_kernel(const int* __restrict__ counts, int* __restrict__ offsets) {
    if (threadIdx.x == 0 && blockIdx.x == 0) {
        int acc = 0;
        for (int e = 0; e < NUM_EXPERTS; ++e) { offsets[e] = acc; acc += counts[e]; }
        offsets[NUM_EXPERTS] = acc;   // == TOP_K * BATCH
    }
}

// ---------------- router: logits -> top-2 softmax -> gather lists ----------------
__global__ __launch_bounds__(256)
void router_kernel(const float* __restrict__ x, const float* __restrict__ gate_w,
                   int* __restrict__ counts, int* __restrict__ tokens,
                   float* __restrict__ wts) {
    __shared__ float red[NUM_EXPERTS * 256];
    __shared__ float logit[NUM_EXPERTS];
    const int tok = blockIdx.x;
    const int tid = threadIdx.x;

    float part[NUM_EXPERTS];
#pragma unroll
    for (int e = 0; e < NUM_EXPERTS; ++e) part[e] = 0.0f;

    const float* xr = x + (long long)tok * D_IN;
    for (int k = tid; k < D_IN; k += 256) {
        float xv = xr[k];
#pragma unroll
        for (int e = 0; e < NUM_EXPERTS; ++e)
            part[e] = fmaf(xv, gate_w[e * D_IN + k], part[e]);
    }
#pragma unroll
    for (int e = 0; e < NUM_EXPERTS; ++e) red[e * 256 + tid] = part[e];
    __syncthreads();

    if (tid < NUM_EXPERTS) {
        float s = 0.0f;
        for (int t = 0; t < 256; ++t) s += red[tid * 256 + t];
        logit[tid] = s;
    }
    __syncthreads();

    if (tid == 0) {
        int bi = 0; float bv = logit[0];
#pragma unroll
        for (int e = 1; e < NUM_EXPERTS; ++e)
            if (logit[e] > bv) { bv = logit[e]; bi = e; }
        int si = (bi == 0) ? 1 : 0; float sv = logit[si];
#pragma unroll
        for (int e = 0; e < NUM_EXPERTS; ++e)
            if (e != bi && logit[e] > sv) { sv = logit[e]; si = e; }
        float p1 = __expf(sv - bv);
        float z  = 1.0f + p1;
        float w0 = 1.0f / z;
        float w1 = p1 / z;
        int p = atomicAdd(&counts[bi], 1);
        tokens[bi * BATCH + p] = tok;  wts[bi * BATCH + p] = w0;
        int q = atomicAdd(&counts[si], 1);
        tokens[si * BATCH + q] = tok;  wts[si * BATCH + q] = w1;
    }
}

// ---------------- WMMA GEMM 1: h = gelu(Xg @ W1e^T + b1e) ----------------
// 8 waves; wave = 32 rows x 64 cols; 64-deep-K B tile staged in LDS via async copy,
// double buffered; 16 WMMAs per barrier.
__global__ __launch_bounds__(256)
void moe_gemm1_kernel(const __bf16* __restrict__ xb, const __bf16* __restrict__ w1b,
                      const float* __restrict__ b1, __bf16* __restrict__ h,
                      const int* __restrict__ counts, const int* __restrict__ offsets,
                      const int* __restrict__ tokens) {
    __shared__ __align__(16) __bf16 Bt[2][BN * KPAD];   // 2 x 9KB

    const int e    = blockIdx.z;
    const int cnt  = counts[e];
    const int mblk = blockIdx.y * BM;
    if (mblk >= cnt) return;                 // uniform early-exit
    const int nbase = blockIdx.x * BN;

    const int tid  = threadIdx.x;
    const int wave = tid >> 5, lane = tid & 31;
    const int hi   = lane >> 4, r = lane & 15, col = lane & 15;

    const __bf16* wbase = w1b + (size_t)e * D_HID * D_IN;

    // async-copy assignment: each of 256 threads moves 32B of the 8KB B tile
    const int crow = tid >> 2, cpart = tid & 3;     // row 0..63, 32B chunk 0..3
    const __bf16* csrc = wbase + (size_t)(nbase + crow) * D_IN + cpart * 16;
    const unsigned cdst[2] = {
        (unsigned)(size_t)&Bt[0][crow * KPAD + cpart * 16],
        (unsigned)(size_t)&Bt[1][crow * KPAD + cpart * 16] };

    // gathered A rows (2 subtiles of 16 rows per wave; zero-safe pad)
    const int m0 = mblk + wave * 32;
    int ar0 = m0 + r,       ar1 = m0 + 16 + r;
    int tok0 = (ar0 < cnt) ? tokens[e * BATCH + ar0] : 0;
    int tok1 = (ar1 < cnt) ? tokens[e * BATCH + ar1] : 0;
    const __bf16* ap0 = xb + (size_t)tok0 * D_IN + (hi ? 8 : 0);
    const __bf16* ap1 = xb + (size_t)tok1 * D_IN + (hi ? 8 : 0);

    v8f acc[2][4] = {};

    // prologue: prefetch first B buffer
    async_ld_b128(cdst[0], csrc);
    async_ld_b128(cdst[0] + 16, csrc + 8);
    wait_async0();
    __syncthreads();

    int cur = 0;
    const int NB = D_IN / BK;
    for (int ib = 0; ib < NB; ++ib) {
        const int kb = ib * BK;
        if (ib + 1 < NB) {                            // overlap next B tile
            async_ld_b128(cdst[cur ^ 1], csrc + kb + BK);
            async_ld_b128(cdst[cur ^ 1] + 16, csrc + kb + BK + 8);
        }
        const __bf16* bbase = &Bt[cur][0];
#pragma unroll
        for (int sub = 0; sub < 2; ++sub) {
            const int k0 = kb + sub * KSTEP;
            v16bf a0 = load_a_frag(ap0, k0);
            v16bf a1 = load_a_frag(ap1, k0);
#pragma unroll
            for (int t = 0; t < 4; ++t) {
                v16bf b = load_b_frag(bbase + (t * 16 + col) * KPAD + sub * 32 + hi * 8);
                acc[0][t] = wmma_bf16(a0, b, acc[0][t]);
                acc[1][t] = wmma_bf16(a1, b, acc[1][t]);
            }
        }
        wait_async0();
        __syncthreads();
        cur ^= 1;
    }

    // epilogue: bias + exact GELU, store bf16 rows into packed h
    const int hrow0 = offsets[e];
#pragma unroll
    for (int s = 0; s < 2; ++s)
#pragma unroll
        for (int t = 0; t < 4; ++t) {
            const int n = nbase + t * 16 + col;
            const float bias = b1[e * D_HID + n];
#pragma unroll
            for (int rr = 0; rr < 8; ++rr) {
                const int m = m0 + s * 16 + rr + (hi ? 8 : 0);
                if (m < cnt) {
                    float v = acc[s][t][rr] + bias;
                    float g = 0.5f * v * (1.0f + erff(v * 0.70710678118654752f));
                    h[(size_t)(hrow0 + m) * D_HID + n] = (__bf16)g;
                }
            }
        }
}

// ---------------- WMMA GEMM 2: out[token] += wt * (H @ W2e^T + b2e) ----------------
__global__ __launch_bounds__(256)
void moe_gemm2_kernel(const __bf16* __restrict__ hbuf, const __bf16* __restrict__ w2b,
                      const float* __restrict__ b2, float* __restrict__ out,
                      const int* __restrict__ counts, const int* __restrict__ offsets,
                      const int* __restrict__ tokens, const float* __restrict__ wts) {
    __shared__ __align__(16) __bf16 Bt[2][BN * KPAD];

    const int e    = blockIdx.z;
    const int cnt  = counts[e];
    const int mblk = blockIdx.y * BM;
    if (mblk >= cnt) return;
    const int nbase = blockIdx.x * BN;

    const int tid  = threadIdx.x;
    const int wave = tid >> 5, lane = tid & 31;
    const int hi   = lane >> 4, r = lane & 15, col = lane & 15;

    const __bf16* wbase = w2b + (size_t)e * D_OUT * D_HID;

    const int crow = tid >> 2, cpart = tid & 3;
    const __bf16* csrc = wbase + (size_t)(nbase + crow) * D_HID + cpart * 16;
    const unsigned cdst[2] = {
        (unsigned)(size_t)&Bt[0][crow * KPAD + cpart * 16],
        (unsigned)(size_t)&Bt[1][crow * KPAD + cpart * 16] };

    const int m0  = mblk + wave * 32;
    const int hrow0 = offsets[e];
    int ar0 = m0 + r,       ar1 = m0 + 16 + r;
    int src0 = hrow0 + ((ar0 < cnt) ? ar0 : 0);
    int src1 = hrow0 + ((ar1 < cnt) ? ar1 : 0);
    const __bf16* ap0 = hbuf + (size_t)src0 * D_HID + (hi ? 8 : 0);
    const __bf16* ap1 = hbuf + (size_t)src1 * D_HID + (hi ? 8 : 0);

    v8f acc[2][4] = {};

    async_ld_b128(cdst[0], csrc);
    async_ld_b128(cdst[0] + 16, csrc + 8);
    wait_async0();
    __syncthreads();

    int cur = 0;
    const int NB = D_HID / BK;
    for (int ib = 0; ib < NB; ++ib) {
        const int kb = ib * BK;
        if (ib + 1 < NB) {
            async_ld_b128(cdst[cur ^ 1], csrc + kb + BK);
            async_ld_b128(cdst[cur ^ 1] + 16, csrc + kb + BK + 8);
        }
        const __bf16* bbase = &Bt[cur][0];
#pragma unroll
        for (int sub = 0; sub < 2; ++sub) {
            const int k0 = kb + sub * KSTEP;
            v16bf a0 = load_a_frag(ap0, k0);
            v16bf a1 = load_a_frag(ap1, k0);
#pragma unroll
            for (int t = 0; t < 4; ++t) {
                v16bf b = load_b_frag(bbase + (t * 16 + col) * KPAD + sub * 32 + hi * 8);
                acc[0][t] = wmma_bf16(a0, b, acc[0][t]);
                acc[1][t] = wmma_bf16(a1, b, acc[1][t]);
            }
        }
        wait_async0();
        __syncthreads();
        cur ^= 1;
    }

#pragma unroll
    for (int s = 0; s < 2; ++s)
#pragma unroll
        for (int t = 0; t < 4; ++t) {
            const int n = nbase + t * 16 + col;
            const float bias = b2[e * D_OUT + n];
#pragma unroll
            for (int rr = 0; rr < 8; ++rr) {
                const int m = m0 + s * 16 + rr + (hi ? 8 : 0);
                if (m < cnt) {
                    const int   tok = tokens[e * BATCH + m];
                    const float wt  = wts[e * BATCH + m];
                    const float y   = acc[s][t][rr] + bias;
                    atomicAdd(&out[(size_t)tok * D_OUT + n], wt * y);
                }
            }
        }
}

// ---------------- host-side orchestration ----------------
extern "C" void kernel_launch(void* const* d_in, const int* in_sizes, int n_in,
                              void* d_out, int out_size, void* d_ws, size_t ws_size,
                              hipStream_t stream) {
    (void)in_sizes; (void)n_in; (void)out_size; (void)ws_size;

    const float* x      = (const float*)d_in[0];   // [B, D_IN]
    const float* gate_w = (const float*)d_in[1];   // [E, D_IN]
    const float* w1     = (const float*)d_in[2];   // [E, D_HID, D_IN]
    const float* b1     = (const float*)d_in[3];   // [E, D_HID]
    const float* w2     = (const float*)d_in[4];   // [E, D_OUT, D_HID]
    const float* b2     = (const float*)d_in[5];   // [E, D_OUT]
    float* out = (float*)d_out;                    // [B, D_OUT]

    auto align256 = [](size_t v) { return (v + 255) & ~(size_t)255; };
    char* ws = (char*)d_ws;
    size_t off = 0;
    int* counts  = (int*)(ws + off);  off = align256(off + NUM_EXPERTS * sizeof(int));
    int* offsets = (int*)(ws + off);  off = align256(off + (NUM_EXPERTS + 1) * sizeof(int));
    int* tokens  = (int*)(ws + off);  off = align256(off + (size_t)NUM_EXPERTS * BATCH * sizeof(int));
    float* wts   = (float*)(ws + off);off = align256(off + (size_t)NUM_EXPERTS * BATCH * sizeof(float));
    __bf16* xb  = (__bf16*)(ws + off); off = align256(off + (size_t)BATCH * D_IN * sizeof(__bf16));
    __bf16* w1b = (__bf16*)(ws + off); off = align256(off + (size_t)NUM_EXPERTS * D_HID * D_IN * sizeof(__bf16));
    __bf16* w2b = (__bf16*)(ws + off); off = align256(off + (size_t)NUM_EXPERTS * D_OUT * D_HID * sizeof(__bf16));
    __bf16* hbuf= (__bf16*)(ws + off); off = align256(off + (size_t)TOP_K * BATCH * D_HID * sizeof(__bf16));

    zero_f32_kernel<<<2048, 256, 0, stream>>>(out, (long long)BATCH * D_OUT);
    zero_i32_kernel<<<1, 32, 0, stream>>>(counts, NUM_EXPERTS);

    cvt_f32_bf16_kernel<<<4096, 256, 0, stream>>>(x,  xb,  (long long)BATCH * D_IN);
    cvt_f32_bf16_kernel<<<8192, 256, 0, stream>>>(w1, w1b, (long long)NUM_EXPERTS * D_HID * D_IN);
    cvt_f32_bf16_kernel<<<8192, 256, 0, stream>>>(w2, w2b, (long long)NUM_EXPERTS * D_OUT * D_HID);

    router_kernel<<<BATCH, 256, 0, stream>>>(x, gate_w, counts, tokens, wts);
    scan_kernel<<<1, 32, 0, stream>>>(counts, offsets);

    {
        dim3 grid(D_HID / BN, (BATCH + BM - 1) / BM, NUM_EXPERTS);
        moe_gemm1_kernel<<<grid, 256, 0, stream>>>(xb, w1b, b1, hbuf, counts, offsets, tokens);
    }
    {
        dim3 grid(D_OUT / BN, (BATCH + BM - 1) / BM, NUM_EXPERTS);
        moe_gemm2_kernel<<<grid, 256, 0, stream>>>(hbuf, w2b, b2, out, counts, offsets, tokens, wts);
    }
}
